// UserProductAttention_15169824489634
// MI455X (gfx1250) — compile-verified, matrix operand order checked by reading, not run on hardware
//
#include <hip/hip_runtime.h>
#include <hip/hip_bf16.h>

typedef __attribute__((ext_vector_type(16))) __bf16 v16bf;
typedef __attribute__((ext_vector_type(8)))  __bf16 v8bf;
typedef __attribute__((ext_vector_type(4)))  __bf16 v4bf;
typedef __attribute__((ext_vector_type(8)))  float  v8f;

// Problem constants
constexpr int BSZ   = 128;
constexpr int SSZ   = 512;
constexpr int HIDN  = 768;
constexpr int OUTN  = 200;
constexpr int OUTP  = 208;            // padded to 13 * 16
constexpr int NT    = 13;             // N tiles of 16
constexpr int KT    = 24;             // K steps of 32 (768/32)
constexpr int NSENT = 16;
constexpr int LSTR  = 776;            // LDS row stride (768 + 8) to break bank conflicts

// Workspace layout (bytes)
constexpr size_t FRAG_ELEMS  = (size_t)NT * KT * 32 * 16;     // 159744 bf16 per array
constexpr size_t OFF_FRAG_HI = 0;
constexpr size_t OFF_FRAG_LO = FRAG_ELEMS * 2;
constexpr size_t OFF_CBUF    = OFF_FRAG_LO + FRAG_ELEMS * 2;  // 128*208 fp32 follows

// LDS layout (bytes)
constexpr int SM_HHI = 0;
constexpr int SM_HLO = 32 * LSTR * 2;
constexpr int SM_SC  = SM_HLO + 32 * LSTR * 2;
constexpr int SM_AL  = SM_SC + 32 * 4;
constexpr int SMEM_BYTES = SM_AL + 32 * 4;                    // 99584 B < 320 KB/WGP

__device__ __forceinline__ v16bf ld_frag16(const __bf16* p0, const __bf16* p1) {
    v8bf a = *(const v8bf*)p0;
    v8bf b = *(const v8bf*)p1;
    return __builtin_shufflevector(a, b, 0,1,2,3,4,5,6,7,8,9,10,11,12,13,14,15);
}

// ---------------------------------------------------------------------------
// Prep 1: split Wh (768x200 fp32) into bf16 hi/lo, stored in WMMA B-fragment
// order: frag[((nt*KT + ks)*32 + lane)*16 + e].
// 16-bit operand lane layout (ISA 7.12.2): lanes 0-15 hold K={0..7,16..23},
// lanes 16-31 hold K={8..15,24..31}; N = lane&15.
// ---------------------------------------------------------------------------
__global__ __launch_bounds__(512) void upa_prep_frag(
    const float* __restrict__ Wh, __bf16* __restrict__ fragHi, __bf16* __restrict__ fragLo)
{
    const int blk  = blockIdx.x;          // nt*KT + ks
    const int nt   = blk / KT;
    const int ks   = blk % KT;
    const int t    = threadIdx.x;         // 0..511 = lane*16 + e
    const int lane = t >> 4;
    const int e    = t & 15;
    const int half = lane >> 4;
    const int n    = nt * 16 + (lane & 15);
    const int K    = (e & 7) + ((e >> 3) * 16) + half * 8;
    const int k    = ks * 32 + K;
    const float w  = (n < OUTN) ? Wh[k * OUTN + n] : 0.f;
    const __bf16 hi = (__bf16)w;
    const __bf16 lo = (__bf16)(w - (float)hi);
    const size_t idx = (size_t)blk * 512 + t;
    fragHi[idx] = hi;
    fragLo[idx] = lo;
}

// ---------------------------------------------------------------------------
// Prep 2: c[b,o] = bias[o] + u[b,:]@Wu[:,o] + p[b,:]@Wp[:,o]   (padded to 208)
// ---------------------------------------------------------------------------
__global__ __launch_bounds__(256) void upa_prep_c(
    const float* __restrict__ u, const float* __restrict__ p,
    const float* __restrict__ Wu, const float* __restrict__ Wp,
    const float* __restrict__ bias, float* __restrict__ cbuf)
{
    const int b = blockIdx.x;
    const int o = threadIdx.x;
    if (o >= OUTP) return;
    float acc = 0.f;
    if (o < OUTN) {
        acc = bias[o];
        const float* ub = u + (size_t)b * OUTN;
        const float* pb = p + (size_t)b * OUTN;
        for (int k = 0; k < OUTN; ++k) {
            acc += ub[k] * Wu[k * OUTN + o];
            acc += pb[k] * Wp[k * OUTN + o];
        }
    }
    cbuf[b * OUTP + o] = acc;
}

// ---------------------------------------------------------------------------
// Main fused kernel: one workgroup per (batch, sentence).
//  A: stage 32x768 H window into LDS as bf16 hi/lo (split-fp32)
//  B: Ht = Hwin @ Wh via v_wmma_f32_16x16x32_bf16, split-bf16 3-pass,
//     dual accumulators over M so each Wh fragment is loaded once per wave
//     scores[s] += v[n] * tanh(Ht + c)           (ds_add_f32 reduction)
//  C: segment softmax, then out = alpha-weighted sum of the LDS window
// ---------------------------------------------------------------------------
__global__ __launch_bounds__(256) void upa_main(
    const float* __restrict__ H, const int* __restrict__ so,
    const __bf16* __restrict__ fragHi, const __bf16* __restrict__ fragLo,
    const float* __restrict__ cbuf, const float* __restrict__ vvec,
    float* __restrict__ out)
{
    extern __shared__ __align__(16) unsigned char smem[];
    __bf16* sHhi = (__bf16*)(smem + SM_HHI);
    __bf16* sHlo = (__bf16*)(smem + SM_HLO);
    float*  sSc  = (float*)(smem + SM_SC);
    float*  sAl  = (float*)(smem + SM_AL);

    const int tid = threadIdx.x;
    const int b   = blockIdx.x >> 4;
    const int m   = blockIdx.x & 15;

    const int beg = so[b * (NSENT + 1) + m];
    const int nxt = so[b * (NSENT + 1) + m + 1];
    const bool valid = (beg != -1);
    const int end = (nxt == -1) ? SSZ : (nxt - 1);
    int cnt = valid ? (end - beg) : 0;
    if (cnt < 0) cnt = 0;
    if (cnt > 32) cnt = 32;

    float* op = out + (size_t)(b * NSENT + m) * HIDN;
    if (cnt == 0) {                          // reference: masked-out row -> zeros
        #pragma unroll
        for (int q = 0; q < HIDN / 256; ++q) op[q * 256 + tid] = 0.f;
        return;
    }

    if (tid < 32) sSc[tid] = 0.f;

    // ---- Phase A: global -> LDS, fp32 split into bf16 hi/lo -------------
    for (int q = 0; q < 24; ++q) {
        const int lin = q * 256 + tid;            // float4 id: 0..6143
        const int r   = lin / (HIDN / 4);
        const int c4  = (lin % (HIDN / 4)) * 4;
        int g = beg + r; if (g > SSZ - 1) g = SSZ - 1;
        const float4 hv = *(const float4*)(H + ((size_t)b * SSZ + g) * HIDN + c4);
        #pragma unroll
        for (int i = 0; i < 4; ++i) {
            const float x = (&hv.x)[i];
            const __bf16 hi = (__bf16)x;
            sHhi[r * LSTR + c4 + i] = hi;
            sHlo[r * LSTR + c4 + i] = (__bf16)(x - (float)hi);
        }
    }
    __syncthreads();

    // ---- Phase B: WMMA GEMM (32 x 208 x 768), split-bf16 3-pass ---------
    const int wave = tid >> 5;
    const int lane = tid & 31;
    const int half = lane >> 4;
    const int lm   = lane & 15;

    for (int nt = wave; nt < NT; nt += 8) {
        v8f acc0 = {};                          // rows 0..15  (mt=0)
        v8f acc1 = {};                          // rows 16..31 (mt=1)
        const int koff = half ? 8 : 0;
        const __bf16* aHi0 = sHhi + lm * LSTR + koff;
        const __bf16* aLo0 = sHlo + lm * LSTR + koff;
        const __bf16* aHi1 = aHi0 + 16 * LSTR;
        const __bf16* aLo1 = aLo0 + 16 * LSTR;
        const __bf16* bHiBase = fragHi + ((size_t)nt * KT * 32 + lane) * 16;
        const __bf16* bLoBase = fragLo + ((size_t)nt * KT * 32 + lane) * 16;

        for (int ks = 0; ks < KT; ++ks) {
            const int k0 = ks * 32;
            const v16bf ahi0 = ld_frag16(aHi0 + k0, aHi0 + k0 + 16);
            const v16bf alo0 = ld_frag16(aLo0 + k0, aLo0 + k0 + 16);
            const v16bf ahi1 = ld_frag16(aHi1 + k0, aHi1 + k0 + 16);
            const v16bf alo1 = ld_frag16(aLo1 + k0, aLo1 + k0 + 16);
            const __bf16* pbh = bHiBase + (size_t)ks * 512;
            const __bf16* pbl = bLoBase + (size_t)ks * 512;
            const v16bf bhi = ld_frag16(pbh, pbh + 8);
            const v16bf blo = ld_frag16(pbl, pbl + 8);
            if (ks + 1 < KT) {                  // stream next fragments into cache
                __builtin_prefetch(pbh + 512, 0, 1);
                __builtin_prefetch(pbl + 512, 0, 1);
            }
            acc0 = __builtin_amdgcn_wmma_f32_16x16x32_bf16(false, ahi0, false, bhi,
                                                           (short)0, acc0, false, false);
            acc1 = __builtin_amdgcn_wmma_f32_16x16x32_bf16(false, ahi1, false, bhi,
                                                           (short)0, acc1, false, false);
            acc0 = __builtin_amdgcn_wmma_f32_16x16x32_bf16(false, ahi0, false, blo,
                                                           (short)0, acc0, false, false);
            acc1 = __builtin_amdgcn_wmma_f32_16x16x32_bf16(false, ahi1, false, blo,
                                                           (short)0, acc1, false, false);
            acc0 = __builtin_amdgcn_wmma_f32_16x16x32_bf16(false, alo0, false, bhi,
                                                           (short)0, acc0, false, false);
            acc1 = __builtin_amdgcn_wmma_f32_16x16x32_bf16(false, alo1, false, bhi,
                                                           (short)0, acc1, false, false);
        }

        // epilogue: scores[s] += v[n] * tanh(Ht[s,n] + c[b,n])
        const int n   = nt * 16 + lm;
        const float cn = cbuf[b * OUTP + n];
        const float vn = (n < OUTN) ? vvec[n] : 0.f;
        #pragma unroll
        for (int r = 0; r < 8; ++r) {
            const int sl = r + (half ? 8 : 0);
            atomicAdd(&sSc[sl],      vn * tanhf(acc0[r] + cn));
            atomicAdd(&sSc[sl + 16], vn * tanhf(acc1[r] + cn));
        }
    }
    __syncthreads();

    // ---- segment softmax over <=32 scores -------------------------------
    if (tid == 0) {
        float mx = -3.4e38f;
        for (int j = 0; j < cnt; ++j) mx = fmaxf(mx, sSc[j]);
        float dn = 0.f;
        for (int j = 0; j < 32; ++j) {
            const float w = (j < cnt) ? expf(sSc[j] - mx) : 0.f;
            sAl[j] = w;
            dn += w;
        }
        const float inv = (dn > 0.f) ? 1.f / dn : 0.f;
        for (int j = 0; j < 32; ++j) sAl[j] *= inv;
    }
    __syncthreads();

    // ---- Phase C: out[b,m,h] = sum_j alpha[j] * H[beg+j, h] (from LDS) --
    // 4 consecutive h per thread -> ds_load_b64 pairs + float4 store
    if (tid < HIDN / 4) {
        const int h0 = tid * 4;
        float s0 = 0.f, s1 = 0.f, s2 = 0.f, s3 = 0.f;
        for (int j = 0; j < 32; ++j) {
            const float a  = sAl[j];
            const v4bf hi = *(const v4bf*)(sHhi + j * LSTR + h0);
            const v4bf lo = *(const v4bf*)(sHlo + j * LSTR + h0);
            s0 += a * ((float)hi[0] + (float)lo[0]);
            s1 += a * ((float)hi[1] + (float)lo[1]);
            s2 += a * ((float)hi[2] + (float)lo[2]);
            s3 += a * ((float)hi[3] + (float)lo[3]);
        }
        float4 r; r.x = s0; r.y = s1; r.z = s2; r.w = s3;
        *(float4*)(op + h0) = r;
    }
}

// ---------------------------------------------------------------------------
extern "C" void kernel_launch(void* const* d_in, const int* in_sizes, int n_in,
                              void* d_out, int out_size, void* d_ws, size_t ws_size,
                              hipStream_t stream) {
    const float* H  = (const float*)d_in[0];
    const float* u  = (const float*)d_in[1];
    const float* p  = (const float*)d_in[2];
    const float* Wh = (const float*)d_in[3];
    const float* Wu = (const float*)d_in[4];
    const float* Wp = (const float*)d_in[5];
    const float* v  = (const float*)d_in[6];
    const float* bi = (const float*)d_in[7];
    const int*   so = (const int*)d_in[8];
    float* out = (float*)d_out;

    unsigned char* ws = (unsigned char*)d_ws;
    __bf16* fragHi = (__bf16*)(ws + OFF_FRAG_HI);
    __bf16* fragLo = (__bf16*)(ws + OFF_FRAG_LO);
    float*  cbuf   = (float*)(ws + OFF_CBUF);

    upa_prep_frag<<<NT * KT, 512, 0, stream>>>(Wh, fragHi, fragLo);
    upa_prep_c<<<BSZ, 256, 0, stream>>>(u, p, Wu, Wp, bi, cbuf);
    upa_main<<<BSZ * NSENT, 256, SMEM_BYTES, stream>>>(H, so, fragHi, fragLo, cbuf, v, out);
}
